// Attention_71476845740690
// MI455X (gfx1250) — matrix operand order repaired and checked by reading, not compiled, not run on hardware
//
#include <hip/hip_runtime.h>
#include <hip/hip_bf16.h>

typedef __attribute__((ext_vector_type(16))) __bf16 v16bf;
typedef __attribute__((ext_vector_type(8)))  __bf16 v8bf;
typedef __attribute__((ext_vector_type(8)))  float  v8f;

#define MASK_VAL (-10000.0f)

__device__ __forceinline__ v8f wmma_bf16(v16bf a, v16bf b, v8f c) {
  return __builtin_amdgcn_wmma_f32_16x16x32_bf16(false, a, false, b, (short)0, c,
                                                 false, false);
}

__device__ __forceinline__ v16bf cat16(v8bf lo, v8bf hi) {
  return __builtin_shufflevector(lo, hi, 0, 1, 2, 3, 4, 5, 6, 7,
                                 8, 9, 10, 11, 12, 13, 14, 15);
}

// A fragment: 16x32 bf16 tile from row-major A (leading dim lda, 16B-aligned rows).
// lanes 0-15 -> M=lane, K 0..7 & 16..23; lanes 16-31 -> M=lane-16, K 8..15 & 24..31.
__device__ __forceinline__ v16bf load_a_frag(const __bf16* A, int lda, int row0, int k0) {
  const int lane = __lane_id();
  const int m    = lane & 15;
  const int kb   = (lane & 16) ? 8 : 0;
  const v8bf* p = (const v8bf*)(A + (long)(row0 + m) * lda + k0 + kb);
  return cat16(p[0], p[2]);  // K kb..kb+7 and kb+16..kb+23
}

// B fragment (32x16, B[K][N]) sourced from row-major Bt[N][K] (ld ldt): B[k][n]=Bt[n][k].
// lanes 0-15: N=lane, K 0..7 & 16..23; lanes 16-31: N=lane-16, K 8..15 & 24..31.
__device__ __forceinline__ v16bf load_bt_frag(const __bf16* Bt, int ldt, int k0, int col0) {
  const int lane = __lane_id();
  const int nn   = lane & 15;
  const int kb   = (lane & 16) ? 8 : 0;
  const v8bf* p = (const v8bf*)(Bt + (long)(col0 + nn) * ldt + k0 + kb);
  return cat16(p[0], p[2]);
}

// ---------------- fp32 -> bf16 conversion (plain) ----------------
__global__ void f32_to_bf16_kernel(const float* __restrict__ in,
                                   __bf16* __restrict__ out, long n) {
  long i = (long)blockIdx.x * blockDim.x + threadIdx.x;
  const long stride = (long)gridDim.x * blockDim.x;
  for (; i < n; i += stride) out[i] = (__bf16)in[i];
}

// ---------------- fp32 -> bf16 conversion with transpose ----------------
// in: [R][C] row-major fp32; out: [C][R] row-major bf16
__global__ void f32_to_bf16_t_kernel(const float* __restrict__ in,
                                     __bf16* __restrict__ out, int R, int C) {
  long i = (long)blockIdx.x * blockDim.x + threadIdx.x;
  const long total = (long)R * C;
  const long stride = (long)gridDim.x * blockDim.x;
  for (; i < total; i += stride) {
    const int r = (int)(i / C);
    const int c = (int)(i % C);
    out[(long)c * R + r] = (__bf16)in[i];
  }
}

// ---------------- bf16 GEMM: C = A[MxK] * B[KxN], B given transposed Bt[NxK] --------
// One wave computes a 64x64 tile: 4x4 WMMA micro-tile, fp32 accumulate.
// MODE 0: bf16 C (ld ldc). MODE 1: fp32 C + bias (ld ldc).
// MODE 2: bf16 C stored transposed per (batch,head): vt[(bi*1024 + col)*2048 + n].
template <int MODE>
__global__ void __launch_bounds__(256)
gemm_bt_kernel(const __bf16* __restrict__ A, const __bf16* __restrict__ Bt,
               void* __restrict__ Cptr, const float* __restrict__ bias,
               int M, int N, int K, int lda, int ldt, int ldc) {
  const int wid     = blockIdx.x * (blockDim.x >> 5) + (threadIdx.x >> 5);
  const int tiles_n = N >> 6;
  const int row0 = (wid / tiles_n) << 6;
  const int col0 = (wid % tiles_n) << 6;

  v8f acc[4][4];
#pragma unroll
  for (int i = 0; i < 4; ++i)
#pragma unroll
    for (int j = 0; j < 4; ++j) acc[i][j] = (v8f){};

  const int lane = __lane_id();
  const int nn   = lane & 15;
  const int mofs = (lane & 16) ? 8 : 0;

  for (int k0 = 0; k0 < K; k0 += 32) {
    if (k0 + 64 < K) {  // prefetch next K-slab (global_prefetch_b8)
      __builtin_prefetch(A + (long)(row0 + nn) * lda + k0 + 64, 0, 3);
      __builtin_prefetch(Bt + (long)(col0 + nn) * ldt + k0 + 64, 0, 3);
    }
    v16bf a[4], b[4];
#pragma unroll
    for (int i = 0; i < 4; ++i) a[i] = load_a_frag(A, lda, row0 + 16 * i, k0);
#pragma unroll
    for (int j = 0; j < 4; ++j) b[j] = load_bt_frag(Bt, ldt, k0, col0 + 16 * j);
#pragma unroll
    for (int i = 0; i < 4; ++i)
#pragma unroll
      for (int j = 0; j < 4; ++j) acc[i][j] = wmma_bf16(a[i], b[j], acc[i][j]);
  }

  if (MODE == 0) {
    __bf16* C = (__bf16*)Cptr;
#pragma unroll
    for (int i = 0; i < 4; ++i)
#pragma unroll
      for (int j = 0; j < 4; ++j)
#pragma unroll
        for (int r = 0; r < 8; ++r)
          C[(long)(row0 + 16 * i + mofs + r) * ldc + col0 + 16 * j + nn] =
              (__bf16)acc[i][j][r];
  } else if (MODE == 1) {
    float* C = (float*)Cptr;
#pragma unroll
    for (int j = 0; j < 4; ++j) {
      const float bv = bias[col0 + 16 * j + nn];
#pragma unroll
      for (int i = 0; i < 4; ++i)
#pragma unroll
        for (int r = 0; r < 8; ++r)
          C[(long)(row0 + 16 * i + mofs + r) * ldc + col0 + 16 * j + nn] =
              acc[i][j][r] + bv;
    }
  } else {  // MODE 2: transposed per (b,h) store; row0..row0+63 within one batch
    __bf16* vt = (__bf16*)Cptr;
    const int bi = row0 >> 11;        // 2048 tokens per batch
    const int nb = row0 & 2047;
#pragma unroll
    for (int i = 0; i < 4; ++i)
#pragma unroll
      for (int j = 0; j < 4; ++j) {
        const long base = ((long)bi * 1024 + col0 + 16 * j + nn) * 2048 + nb + 16 * i + mofs;
#pragma unroll
        for (int r = 0; r < 8; ++r) vt[base + r] = (__bf16)acc[i][j][r];
      }
  }
}

// ---------------- causal flash attention (S^T orientation) ----------------
// Computes S^T = K * Q^T per 32-key block so that:
//  * query index lands on the lane axis -> softmax stats are in-lane + one shfl_xor(16)
//  * S^T's C-layout IS the A-fragment layout of P[16q x 32k] -> no LDS round-trip
// q/kbuf: [b*2048, 1024] bf16 (head h at col h*64); vt: per (b,h) V^T [64][2048] bf16.
__global__ void __launch_bounds__(128)
flash_attn_kernel(const __bf16* __restrict__ Q, const __bf16* __restrict__ Kb,
                  const __bf16* __restrict__ Vt, __bf16* __restrict__ Oattn) {
  const int Nctx = 2048, DIM = 1024, D = 64, H = 16;
  const float scale = 0.03125f;  // 1024^-0.5 exactly

  const int wib    = threadIdx.x >> 5;
  const int wid    = blockIdx.x * 4 + wib;
  const int qtiles = Nctx / 16;  // 128
  const int qt = wid % qtiles;
  const int bh = wid / qtiles;
  const int hi = bh % H;
  const int bi = bh / H;

  const int lane = __lane_id();
  const int nn   = lane & 15;
  const int mofs = (lane & 16) ? 8 : 0;  // C-layout M offset == A-frag K-half offset

  const __bf16* Qp = Q  + (long)bi * Nctx * DIM + hi * D;
  const __bf16* Kp = Kb + (long)bi * Nctx * DIM + hi * D;
  const __bf16* Vp = Vt + ((long)bi * H + hi) * (long)D * Nctx;  // [64][2048]

  const int q0 = qt * 16;
  const int qn = q0 + nn;  // the query whose stats this lane tracks

  // Q^T B-fragments (d on the K axis), hoisted out of the key loop
  const v16bf qb0 = load_bt_frag(Qp, DIM, 0,  q0);
  const v16bf qb1 = load_bt_frag(Qp, DIM, 32, q0);

  v8f o0 = {}, o1 = {}, o2 = {}, o3 = {};  // O[query][d] C-layout
  float m_run = -3.0e38f, l_run = 0.0f;

  const int kend = q0 + 15;  // causal: keys <= query row
  for (int kb = 0; kb <= kend; kb += 32) {
    if (kb + 32 <= kend)
      __builtin_prefetch(Kp + (long)(kb + 32 + nn) * DIM, 0, 3);

    // S^T tiles: [key][query]; keys kb..kb+15 (st0), kb+16..kb+31 (st1)
    v8f st0 = {}, st1 = {};
    st0 = wmma_bf16(load_a_frag(Kp, DIM, kb,      0),  qb0, st0);
    st0 = wmma_bf16(load_a_frag(Kp, DIM, kb,      32), qb1, st0);
    st1 = wmma_bf16(load_a_frag(Kp, DIM, kb + 16, 0),  qb0, st1);
    st1 = wmma_bf16(load_a_frag(Kp, DIM, kb + 16, 32), qb1, st1);

    // scale + causal mask; st0 elem r <-> key kb+mofs+r, st1 <-> +16
    float e0[8], e1[8];
#pragma unroll
    for (int r = 0; r < 8; ++r) {
      const int k0i = kb + mofs + r;
      e0[r] = (k0i      <= qn) ? st0[r] * scale : MASK_VAL;
      e1[r] = (k0i + 16 <= qn) ? st1[r] * scale : MASK_VAL;
    }

    // block max for query qn: in-lane over 16 values, then merge lane halves
    float mx = fmaxf(e0[0], e1[0]);
#pragma unroll
    for (int r = 1; r < 8; ++r) mx = fmaxf(mx, fmaxf(e0[r], e1[r]));
    mx = fmaxf(mx, __shfl_xor(mx, 16, 32));

    const float mnew  = fmaxf(m_run, mx);
    const float alpha = __expf(m_run - mnew);
    m_run = mnew;

    // exponentiate; pack directly into the P A-fragment (layout coincidence)
    v16bf pa;
    float rs = 0.0f;
#pragma unroll
    for (int r = 0; r < 8; ++r) {
      const float p0 = __expf(e0[r] - mnew);
      const float p1 = __expf(e1[r] - mnew);
      rs += p0 + p1;
      pa[r]     = (__bf16)p0;
      pa[8 + r] = (__bf16)p1;
    }
    rs += __shfl_xor(rs, 16, 32);
    l_run = l_run * alpha + rs;

    // broadcast alpha from lane axis to O-row axis, rescale accumulators
#pragma unroll
    for (int r = 0; r < 8; ++r) {
      const float ar = __shfl(alpha, mofs + r, 32);
      o0[r] *= ar; o1[r] *= ar; o2[r] *= ar; o3[r] *= ar;
    }

    // O += P[16x32] * V[32x64]; V^T gives contiguous B-fragments
    o0 = wmma_bf16(pa, load_bt_frag(Vp, Nctx, kb, 0),  o0);
    o1 = wmma_bf16(pa, load_bt_frag(Vp, Nctx, kb, 16), o1);
    o2 = wmma_bf16(pa, load_bt_frag(Vp, Nctx, kb, 32), o2);
    o3 = wmma_bf16(pa, load_bt_frag(Vp, Nctx, kb, 48), o3);
  }

  __bf16* Op = Oattn + (long)bi * Nctx * DIM + hi * D;
#pragma unroll
  for (int r = 0; r < 8; ++r) {
    const float inv  = 1.0f / __shfl(l_run, mofs + r, 32);
    const long  rowo = (long)(q0 + mofs + r) * DIM;
    Op[rowo + 0  + nn] = (__bf16)(o0[r] * inv);
    Op[rowo + 16 + nn] = (__bf16)(o1[r] * inv);
    Op[rowo + 32 + nn] = (__bf16)(o2[r] * inv);
    Op[rowo + 48 + nn] = (__bf16)(o3[r] * inv);
  }
}

extern "C" void kernel_launch(void* const* d_in, const int* in_sizes, int n_in,
                              void* d_out, int out_size, void* d_ws, size_t ws_size,
                              hipStream_t stream) {
  (void)in_sizes; (void)n_in; (void)out_size; (void)ws_size;
  const float* x    = (const float*)d_in[0];
  const float* Wq   = (const float*)d_in[1];
  const float* Wkv  = (const float*)d_in[2];
  const float* Wout = (const float*)d_in[3];
  const float* bout = (const float*)d_in[4];
  float* out = (float*)d_out;

  const long BN  = 4L * 2048;  // 8192 token rows
  const long DIM = 1024, KVD = 2048;

  char* ws = (char*)d_ws;
  __bf16* xb    = (__bf16*)ws; ws += BN * DIM * 2;     // 16 MiB
  __bf16* wqt   = (__bf16*)ws; ws += DIM * DIM * 2;    //  2 MiB  Wq^T   [1024][1024]
  __bf16* wkvt  = (__bf16*)ws; ws += DIM * KVD * 2;    //  4 MiB  Wkv^T  [2048][1024]
  __bf16* woutt = (__bf16*)ws; ws += DIM * DIM * 2;    //  2 MiB  Wout^T [1024][1024]
  __bf16* qb    = (__bf16*)ws; ws += BN * DIM * 2;     // 16 MiB
  __bf16* kbuf  = (__bf16*)ws; ws += BN * DIM * 2;     // 16 MiB
  __bf16* vt    = (__bf16*)ws; ws += BN * DIM * 2;     // 16 MiB  per (b,h): [64][2048]
  __bf16* ab    = (__bf16*)ws; ws += BN * DIM * 2;     // 16 MiB

  const __bf16* wkt = wkvt;                        // rows 0..1023   = Wk^T
  const __bf16* wvt = wkvt + (long)DIM * DIM;      // rows 1024..2047 = Wv^T

  f32_to_bf16_kernel<<<1024, 256, 0, stream>>>(x, xb, BN * DIM);
  f32_to_bf16_t_kernel<<<256, 256, 0, stream>>>(Wq, wqt, (int)DIM, (int)DIM);
  f32_to_bf16_t_kernel<<<512, 256, 0, stream>>>(Wkv, wkvt, (int)DIM, (int)KVD);
  f32_to_bf16_t_kernel<<<256, 256, 0, stream>>>(Wout, woutt, (int)DIM, (int)DIM);

  const int gemm_blocks = (int)((BN / 64) * (DIM / 64) / 8);  // 256 blocks, 8 waves each

  // Q = x @ Wq
  gemm_bt_kernel<0><<<gemm_blocks, 256, 0, stream>>>(xb, wqt, qb, nullptr,
      (int)BN, (int)DIM, (int)DIM, (int)DIM, (int)DIM, (int)DIM);
  // K = x @ Wk
  gemm_bt_kernel<0><<<gemm_blocks, 256, 0, stream>>>(xb, wkt, kbuf, nullptr,
      (int)BN, (int)DIM, (int)DIM, (int)DIM, (int)DIM, (int)DIM);
  // V = x @ Wv, stored transposed per (b,h) into vt
  gemm_bt_kernel<2><<<gemm_blocks, 256, 0, stream>>>(xb, wvt, vt, nullptr,
      (int)BN, (int)DIM, (int)DIM, (int)DIM, (int)DIM, 0);
  // causal flash attention (4*16*128 = 8192 waves)
  flash_attn_kernel<<<2048, 128, 0, stream>>>(qb, kbuf, vt, ab);
  // out = attn @ Wout + bout (fp32)
  gemm_bt_kernel<1><<<gemm_blocks, 256, 0, stream>>>(ab, woutt, out, bout,
      (int)BN, (int)DIM, (int)DIM, (int)DIM, (int)DIM, (int)DIM);
}
